// Attention_41377714930037
// MI455X (gfx1250) — matrix operand order repaired and checked by reading, not compiled, not run on hardware
//
#include <hip/hip_runtime.h>
#include <hip/hip_bf16.h>
#include <math.h>

typedef __attribute__((ext_vector_type(16))) __bf16 v16bf;
typedef __attribute__((ext_vector_type(8)))  __bf16 v8bf;
typedef __attribute__((ext_vector_type(8)))  float  v8f;

#define HID     256
#define SLEN    8192
#define NB      32
#define THREADS 512      // 16 waves (wave32)
#define STILES  16       // s-tiles of 16 rows per workgroup
#define WGPB    (SLEN / (16 * STILES))   // 32 workgroups per batch

__device__ __forceinline__ float fast_tanh(float x) {
#if __has_builtin(__builtin_amdgcn_tanhf)
    return __builtin_amdgcn_tanhf(x);        // v_tanh_f32 (CDNA5 TRANS op)
#elif __has_builtin(__builtin_amdgcn_tanh_f32)
    return __builtin_amdgcn_tanh_f32(x);
#else
    return tanhf(x);
#endif
}

// Butterfly sum over each 16-lane half of the wave, entirely on the VALU pipe
// (DPP8 for xor 1/2/4, DPP16 row_ror:8 for the cross-8 step) so it does not
// contend with other waves' ds_load_b128 traffic on the LDS pipe.
#define DPP8_SEL(a,b,c,d,e,f,g,h) \
    ((a)|((b)<<3)|((c)<<6)|((d)<<9)|((e)<<12)|((f)<<15)|((g)<<18)|((h)<<21))

__device__ __forceinline__ float lane16_sum(float x) {
#if __has_builtin(__builtin_amdgcn_mov_dpp8) && __has_builtin(__builtin_amdgcn_mov_dpp)
    x += __int_as_float(__builtin_amdgcn_mov_dpp8(__float_as_int(x),
                         DPP8_SEL(1,0,3,2,5,4,7,6)));                 // xor 1
    x += __int_as_float(__builtin_amdgcn_mov_dpp8(__float_as_int(x),
                         DPP8_SEL(2,3,0,1,6,7,4,5)));                 // xor 2
    x += __int_as_float(__builtin_amdgcn_mov_dpp8(__float_as_int(x),
                         DPP8_SEL(4,5,6,7,0,1,2,3)));                 // xor 4
    x += __int_as_float(__builtin_amdgcn_mov_dpp(__float_as_int(x),
                         0x128 /*row_ror:8*/, 0xf, 0xf, true));       // cross-8
#else
    x += __shfl_xor(x, 1);
    x += __shfl_xor(x, 2);
    x += __shfl_xor(x, 4);
    x += __shfl_xor(x, 8);
#endif
    return x;
}

__device__ __forceinline__ v8bf cvt8(float4 a, float4 b) {
    v8bf r;
    r[0] = (__bf16)a.x; r[1] = (__bf16)a.y; r[2] = (__bf16)a.z; r[3] = (__bf16)a.w;
    r[4] = (__bf16)b.x; r[5] = (__bf16)b.y; r[6] = (__bf16)b.z; r[7] = (__bf16)b.w;
    return r;
}

// -------- Pass 1: fused GEMM(bf16 WMMA) + tanh + dot(v) -> logits ------------
__global__ __launch_bounds__(THREADS)
void attn_pass1(const float* __restrict__ hidden, const float* __restrict__ enc,
                const float* __restrict__ W, const float* __restrict__ bias,
                const float* __restrict__ v, float* __restrict__ logits)
{
    // Double-buffered A tile: 16 s-rows x 256 h, bf16, stride 264 -> conflict-free ds_load_b128
    __shared__ __align__(16) __bf16 ldsA[2][16][264];
    __shared__ float ldsH[HID];       // hidden @ Wh^T + bias (per-batch constant over s)
    __shared__ float ldsLogit[16];

    const int tid  = threadIdx.x;
    const int lane = tid & 31;
    const int wv   = tid >> 5;        // wave id 0..15 -> k-column tile
    const int b    = blockIdx.y;
    const int half = (lane >= 16) ? 1 : 0;
    const int ln   = lane & 15;

    // hconst[k] = bias[k] + sum_h hidden[b,h] * W[k,h]   (Wh = W[:, :256], indexed [k,h])
    if (tid < HID) {
        const float* hp = hidden + (size_t)b * HID;
        const float* wp = W + (size_t)tid * (2 * HID);
        float acc = bias[tid];
        #pragma unroll 8
        for (int h = 0; h < HID; ++h) acc += hp[h] * wp[h];
        ldsH[tid] = acc;
    }

    // B fragments: We^T columns for this wave's 16 k-outputs, all 8 K-steps,
    // register-resident (contiguous k order: lane holds N=lane&15, K=j+16*(lane>=16)).
    const int kcol = wv * 16 + ln;
    v16bf Bf[8];
    #pragma unroll
    for (int t = 0; t < 8; ++t) {
        const float* wp = W + (size_t)kcol * (2 * HID) + HID + 32 * t + 16 * half;
        float4 q0 = *(const float4*)(wp + 0);
        float4 q1 = *(const float4*)(wp + 4);
        float4 q2 = *(const float4*)(wp + 8);
        float4 q3 = *(const float4*)(wp + 12);
        v8bf lo = cvt8(q0, q1), hi = cvt8(q2, q3);
        Bf[t] = __builtin_shufflevector(lo, hi, 0,1,2,3,4,5,6,7,8,9,10,11,12,13,14,15);
    }

    __syncthreads();
    const float hc = ldsH[kcol];
    const float vk = v[kcol];

    const int srow = tid >> 5;             // staging: thread -> (row, 8-float run)
    const int scol = (tid & 31) * 8;
    const float* apBase =
        enc + (size_t)b * SLEN * HID
            + (size_t)(blockIdx.x * STILES * 16 + srow) * HID + scol;

    // software pipeline: global loads for tile st+1 issue under tile st's WMMAs
    float4 a0 = *(const float4*)(apBase);
    float4 a1 = *(const float4*)(apBase + 4);

    for (int st = 0; st < STILES; ++st) {
        const int buf = st & 1;
        const int s0  = (blockIdx.x * STILES + st) * 16;

        *(v8bf*)&ldsA[buf][srow][scol] = cvt8(a0, a1);
        if (tid < 16) ldsLogit[tid] = 0.0f;
        __syncthreads();                      // A(buf) staged, logit cleared

        if (st + 1 < STILES) {                // prefetch next tile into registers
            const float* ap = apBase + (size_t)(st + 1) * 16 * HID;
            a0 = *(const float4*)(ap);
            a1 = *(const float4*)(ap + 4);
        }

        // A layout: lane row M=lane&15; element j -> K=16*half+j
        const __bf16* arp = &ldsA[buf][ln][16 * half];
        v16bf Aall[8];
        #pragma unroll
        for (int t = 0; t < 8; ++t) {
            v8bf lo = *(const v8bf*)(arp + 32 * t);
            v8bf hi = *(const v8bf*)(arp + 32 * t + 8);
            Aall[t] = __builtin_shufflevector(lo, hi,
                          0,1,2,3,4,5,6,7,8,9,10,11,12,13,14,15);
        }
        // Two independent accumulator chains -> back-to-back XDL pipelining and
        // overlap of fragment loads with in-flight WMMAs.
        v8f acc0, acc1;
        #pragma unroll
        for (int r = 0; r < 8; ++r) { acc0[r] = 0.0f; acc1[r] = 0.0f; }
        #pragma unroll
        for (int t = 0; t < 8; t += 2) {
            acc0 = __builtin_amdgcn_wmma_f32_16x16x32_bf16(
                       false, Aall[t],     false, Bf[t],     (short)0, acc0, false, false);
            acc1 = __builtin_amdgcn_wmma_f32_16x16x32_bf16(
                       false, Aall[t + 1], false, Bf[t + 1], (short)0, acc1, false, false);
        }

        // Epilogue: D element (M=r+8*half, N=ln). partial logit contribution.
        float part[8];
        #pragma unroll
        for (int r = 0; r < 8; ++r)
            part[r] = fast_tanh(acc0[r] + acc1[r] + hc) * vk;
        #pragma unroll
        for (int r = 0; r < 8; ++r) part[r] = lane16_sum(part[r]);
        if (ln == 0) {
            #pragma unroll
            for (int r = 0; r < 8; ++r) atomicAdd(&ldsLogit[half * 8 + r], part[r]);
        }
        __syncthreads();                      // all 16 waves' partials in
        if (tid < 16) logits[(size_t)b * SLEN + s0 + tid] = ldsLogit[tid];
    }
}

// -------- Pass 2: softmax over s=8192, in place ------------------------------
__global__ __launch_bounds__(256)
void attn_pass2(float* __restrict__ attn)
{
    __shared__ float red[256];
    const int tid = threadIdx.x;
    float* p = attn + (size_t)blockIdx.x * SLEN;
    float vals[32];
    float m = -__builtin_inff();
    #pragma unroll
    for (int i = 0; i < 32; ++i) { vals[i] = p[tid + 256 * i]; m = fmaxf(m, vals[i]); }
    red[tid] = m; __syncthreads();
    for (int off = 128; off > 0; off >>= 1) {
        if (tid < off) red[tid] = fmaxf(red[tid], red[tid + off]);
        __syncthreads();
    }
    m = red[0]; __syncthreads();
    float s = 0.0f;
    #pragma unroll
    for (int i = 0; i < 32; ++i) { vals[i] = __expf(vals[i] - m); s += vals[i]; }
    red[tid] = s; __syncthreads();
    for (int off = 128; off > 0; off >>= 1) {
        if (tid < off) red[tid] += red[tid + off];
        __syncthreads();
    }
    const float inv = 1.0f / red[0];
    #pragma unroll
    for (int i = 0; i < 32; ++i) p[tid + 256 * i] = vals[i] * inv;
}

// -------- Pass 3: context partials (bandwidth bound) -------------------------
__global__ __launch_bounds__(256)
void attn_pass3(const float* __restrict__ enc, const float* __restrict__ attn,
                float* __restrict__ partial, int schunk)
{
    const int h = threadIdx.x;
    const int c = blockIdx.x;
    const int b = blockIdx.y;
    const int sBeg = c * schunk;
    const float* ap = attn + (size_t)b * SLEN;
    const float* ep = enc + ((size_t)b * SLEN + sBeg) * HID + h;
    float acc = 0.0f;
    for (int s = sBeg; s < sBeg + schunk; ++s) { acc += ap[s] * (*ep); ep += HID; }
    partial[((size_t)b * gridDim.x + c) * HID + h] = acc;
}

// -------- Pass 4: reduce partials -> context ---------------------------------
__global__ __launch_bounds__(256)
void attn_pass4(const float* __restrict__ partial, float* __restrict__ ctx, int nchunks)
{
    const int h = threadIdx.x;
    const int b = blockIdx.x;
    float acc = 0.0f;
    for (int c = 0; c < nchunks; ++c) acc += partial[((size_t)b * nchunks + c) * HID + h];
    ctx[(size_t)b * HID + h] = acc;
}

extern "C" void kernel_launch(void* const* d_in, const int* in_sizes, int n_in,
                              void* d_out, int out_size, void* d_ws, size_t ws_size,
                              hipStream_t stream)
{
    const float* hidden = (const float*)d_in[0];   // (1,32,256)
    const float* enc    = (const float*)d_in[1];   // (32,8192,256)
    const float* W      = (const float*)d_in[2];   // (256,512)
    const float* bias   = (const float*)d_in[3];   // (256,)
    const float* v      = (const float*)d_in[4];   // (256,)

    float* ctx  = (float*)d_out;                   // (32,256)
    float* attn = ctx + NB * HID;                  // (32,8192), logits then softmax in place

    int nchunks = 16;
    if (ws_size < (size_t)NB * 16 * HID * sizeof(float)) nchunks = 4;
    const int schunk = SLEN / nchunks;
    float* partial = (float*)d_ws;                 // (32, nchunks, 256)

    attn_pass1<<<dim3(WGPB, NB), THREADS, 0, stream>>>(hidden, enc, W, bias, v, attn);
    attn_pass2<<<dim3(NB), 256, 0, stream>>>(attn);
    attn_pass3<<<dim3(nchunks, NB), 256, 0, stream>>>(enc, attn, partial, schunk);
    attn_pass4<<<dim3(NB), 256, 0, stream>>>(partial, ctx, nchunks);
}
